// DynamicMoEBlock_13881334300820
// MI455X (gfx1250) — matrix operand order repaired and checked by reading, not compiled, not run on hardware
//
#include <hip/hip_runtime.h>
#include <math.h>

// ---------------------------------------------------------------------------
// MoE block for MI455X (gfx1250, wave32, WMMA).
// Dense per-expert GEMMs in bf16 WMMA (f32 accumulate), top-2 combine weights
// applied in the second GEMM's epilogue. Experts run sequentially on the
// stream so fp32 accumulation into d_out is deterministic (no atomics).
// ---------------------------------------------------------------------------

typedef __attribute__((ext_vector_type(16))) __bf16 v16bf;
typedef __attribute__((ext_vector_type(8)))  float  v8f;
typedef __attribute__((ext_vector_type(4)))  float  v4f;
typedef __attribute__((ext_vector_type(4)))  unsigned int v4u;

#define T_TOK 4096   // B*S
#define HDIM  1024
#define FDIM  4096
#define NEXP  8
#define COMB_STRIDE 9  // 8 experts + shared

union AFrag { v4u u[2]; v16bf v; };
union BfPair { __bf16 h[2]; unsigned int u; };

__device__ __forceinline__ float gelu_exact(float v) {
    return 0.5f * v * (1.0f + erff(v * 0.70710678118654752f));
}

// ---------------------------------------------------------------------------
// x (fp32) -> xb (bf16), 8 elements per thread
// ---------------------------------------------------------------------------
__global__ __launch_bounds__(256) void moe_convert_bf16(
    const float* __restrict__ x, __bf16* __restrict__ xb, int n) {
    int i = (blockIdx.x * 256 + threadIdx.x) * 8;
    if (i >= n) return;
    v4f a = *(const v4f*)(x + i);
    v4f b = *(const v4f*)(x + i + 4);
    __bf16 o[8];
    o[0] = (__bf16)a[0]; o[1] = (__bf16)a[1]; o[2] = (__bf16)a[2]; o[3] = (__bf16)a[3];
    o[4] = (__bf16)b[0]; o[5] = (__bf16)b[1]; o[6] = (__bf16)b[2]; o[7] = (__bf16)b[3];
    *(v4u*)(xb + i) = *(const v4u*)o;
}

// ---------------------------------------------------------------------------
// Gating: logits = x @ gate_w^T, softmax, top-2, renormalize.
// One thread per token; gate_w (8x1024 fp32 = 32KB) staged in LDS.
// comb[t*9 + e] = renormalized weight (0 if not selected); comb[t*9+8] = 1.
// ---------------------------------------------------------------------------
__global__ __launch_bounds__(256) void moe_gate(
    const float* __restrict__ x, const float* __restrict__ gate_w,
    float* __restrict__ comb) {
    __shared__ float gws[NEXP * HDIM];
    int tid = threadIdx.x;
    for (int i = tid; i < NEXP * HDIM; i += 256) gws[i] = gate_w[i];
    __syncthreads();

    int t = blockIdx.x * 256 + tid;
    const float* xr = x + (size_t)t * HDIM;
    float acc[NEXP];
    #pragma unroll
    for (int e = 0; e < NEXP; ++e) acc[e] = 0.0f;

    for (int h4 = 0; h4 < HDIM / 4; ++h4) {
        v4f xv = *(const v4f*)(xr + h4 * 4);
        #pragma unroll
        for (int c = 0; c < 4; ++c) {
            float xs = xv[c];
            #pragma unroll
            for (int e = 0; e < NEXP; ++e)
                acc[e] += xs * gws[e * HDIM + h4 * 4 + c];
        }
    }

    // top-1 (ties -> lowest index, matching lax.top_k)
    int a0 = 0;
    #pragma unroll
    for (int e = 1; e < NEXP; ++e) if (acc[e] > acc[a0]) a0 = e;
    // top-2
    int a1 = (a0 == 0) ? 1 : 0;
    #pragma unroll
    for (int e = 0; e < NEXP; ++e)
        if (e != a0 && acc[e] > acc[a1]) a1 = e;
    // keep lowest-index-first on exact ties among non-top1
    #pragma unroll
    for (int e = 0; e < NEXP; ++e)
        if (e != a0 && acc[e] == acc[a1] && e < a1) a1 = e;

    float m  = acc[a0];
    float p0 = 1.0f;                 // exp(0)
    float p1 = __expf(acc[a1] - m);
    float inv = 1.0f / (p0 + p1);

    float* cr = comb + (size_t)t * COMB_STRIDE;
    #pragma unroll
    for (int e = 0; e < NEXP; ++e) cr[e] = 0.0f;
    cr[a0] = p0 * inv;
    cr[a1] = p1 * inv;
    cr[NEXP] = 1.0f;                 // shared expert weight
}

// ---------------------------------------------------------------------------
// Tile staging: A tile 128x32 (bf16 source), row-major [m][32] in LDS.
// ---------------------------------------------------------------------------
__device__ __forceinline__ void load_a_tile(
    __bf16* __restrict__ dst, const __bf16* __restrict__ A,
    int m0, int kk, int K, int tid) {
    int row = tid >> 1;
    int seg = (tid & 1) * 16;
    const __bf16* src = A + (size_t)(m0 + row) * K + kk + seg;
    v4u d0 = *(const v4u*)(src);
    v4u d1 = *(const v4u*)(src + 8);
    *(v4u*)(dst + row * 32 + seg)     = d0;
    *(v4u*)(dst + row * 32 + seg + 8) = d1;
}

// B tile 32x128 from fp32 [K,N] row-major, stored k-contiguous per column
// (LDS [n][32 k]). Each thread owns a k-PAIR (two consecutive W rows) for 8
// columns: one v_cvt_pk_bf16_f32 + one ds_store_b32 per column (k-pair packed
// exactly as the B fragment expects: VGPR v holds K=j,j+1, j=2v+8*(lane/16)+8*(v>=4)).
__device__ __forceinline__ void load_b_tile(
    __bf16* __restrict__ dst, const float* __restrict__ W,
    int n0, int kk, int N, int tid) {
    int kp = tid >> 4;            // 0..15  (k = 2*kp, 2*kp+1)
    int nb = (tid & 15) * 8;      // 0..120 step 8
    const float* r0 = W + (size_t)(kk + 2 * kp) * N + n0 + nb;
    const float* r1 = r0 + N;
    #pragma unroll
    for (int s = 0; s < 2; ++s) {
        v4f f0 = *(const v4f*)(r0 + s * 4);
        v4f f1 = *(const v4f*)(r1 + s * 4);
        #pragma unroll
        for (int c = 0; c < 4; ++c) {
            BfPair p;
            p.h[0] = (__bf16)f0[c];
            p.h[1] = (__bf16)f1[c];
            *(unsigned int*)(dst + (nb + s * 4 + c) * 32 + 2 * kp) = p.u;
        }
    }
}

// Prefetch the staging addresses of a future K-tile (emits global_prefetch_b8).
__device__ __forceinline__ void prefetch_tiles(
    const __bf16* __restrict__ A, const float* __restrict__ W,
    int m0, int n0, int kk, int N, int K, int tid) {
    int row = tid >> 1;
    int seg = (tid & 1) * 16;
    __builtin_prefetch(A + (size_t)(m0 + row) * K + kk + seg, 0, 0);
    int kp = tid >> 4;
    int nb = (tid & 15) * 8;
    __builtin_prefetch(W + (size_t)(kk + 2 * kp) * N + n0 + nb, 0, 0);
    __builtin_prefetch(W + (size_t)(kk + 2 * kp + 1) * N + n0 + nb, 0, 0);
}

// Fragment gather per ISA VGPR layouts (16-bit A 16x32; B column-contiguous).
__device__ __forceinline__ void make_frag_a(
    AFrag& a, const __bf16* __restrict__ As, int mb, int lidx, int lhalf) {
    const __bf16* p = As + (mb + lidx) * 32;
    a.u[0] = *(const v4u*)(p + lhalf * 8);
    a.u[1] = *(const v4u*)(p + 16 + lhalf * 8);
}
__device__ __forceinline__ void make_frag_b(
    AFrag& b, const __bf16* __restrict__ Bs, int nb, int lidx, int lhalf) {
    const __bf16* p = Bs + (nb + lidx) * 32;
    b.u[0] = *(const v4u*)(p + lhalf * 8);
    b.u[1] = *(const v4u*)(p + 16 + lhalf * 8);
}

// ---------------------------------------------------------------------------
// GEMM1: Hout[m,n] = gelu(A[m,:] @ W[:,n] + bias[n]), bf16 out.
// A: [M,K] bf16 row-major. W: [K,N] fp32 row-major. Grid: (N/128, M/128).
// 256 threads = 8 waves; wave grid 4(M) x 2(N); each wave: 2x4 16x16 tiles.
// ---------------------------------------------------------------------------
__global__ __launch_bounds__(256) void moe_gemm1(
    const __bf16* __restrict__ A, const float* __restrict__ W,
    const float* __restrict__ bias, __bf16* __restrict__ Hout,
    int N, int K) {
    __shared__ __bf16 As[2][128 * 32];
    __shared__ __bf16 Bs[2][128 * 32];

    int tid = threadIdx.x;
    int n0 = blockIdx.x * 128, m0 = blockIdx.y * 128;
    int w = tid >> 5, lane = tid & 31;
    int wm = w >> 1, wn = w & 1;
    int lidx = lane & 15, lhalf = lane >> 4;

    v8f acc[2][4];
    #pragma unroll
    for (int i = 0; i < 2; ++i)
        #pragma unroll
        for (int j = 0; j < 4; ++j)
            acc[i][j] = (v8f){0.f,0.f,0.f,0.f,0.f,0.f,0.f,0.f};

    load_a_tile(As[0], A, m0, 0, K, tid);
    load_b_tile(Bs[0], W, n0, 0, N, tid);
    prefetch_tiles(A, W, m0, n0, 32, N, K, tid);
    __syncthreads();

    int KT = K >> 5;
    for (int kt = 0; kt < KT; ++kt) {
        int cur = kt & 1;
        if (kt + 1 < KT) {
            load_a_tile(As[cur ^ 1], A, m0, (kt + 1) << 5, K, tid);
            load_b_tile(Bs[cur ^ 1], W, n0, (kt + 1) << 5, N, tid);
        }
        if (kt + 2 < KT)
            prefetch_tiles(A, W, m0, n0, (kt + 2) << 5, N, K, tid);
        AFrag a[2], b[4];
        #pragma unroll
        for (int i = 0; i < 2; ++i)
            make_frag_a(a[i], As[cur], wm * 32 + i * 16, lidx, lhalf);
        #pragma unroll
        for (int j = 0; j < 4; ++j)
            make_frag_b(b[j], Bs[cur], wn * 64 + j * 16, lidx, lhalf);
        #pragma unroll
        for (int i = 0; i < 2; ++i)
            #pragma unroll
            for (int j = 0; j < 4; ++j)
                acc[i][j] = __builtin_amdgcn_wmma_f32_16x16x32_bf16(
                    false, a[i].v, false, b[j].v, (short)0, acc[i][j],
                    false, false);
        __syncthreads();
    }

    // epilogue: C/D layout -> m = r + 8*(lane>=16), n = lane%16
    #pragma unroll
    for (int i = 0; i < 2; ++i) {
        int rm = m0 + wm * 32 + i * 16 + lhalf * 8;
        #pragma unroll
        for (int j = 0; j < 4; ++j) {
            int cn = n0 + wn * 64 + j * 16 + lidx;
            float bv = bias[cn];
            #pragma unroll
            for (int r = 0; r < 8; ++r) {
                float v = acc[i][j][r] + bv;
                Hout[(size_t)(rm + r) * N + cn] = (__bf16)gelu_exact(v);
            }
        }
    }
}

// ---------------------------------------------------------------------------
// GEMM2: out[m,n] += comb[m,col] * (A[m,:] @ W[:,n] + bias[n]), fp32 accum.
// ---------------------------------------------------------------------------
__global__ __launch_bounds__(256) void moe_gemm2(
    const __bf16* __restrict__ A, const float* __restrict__ W,
    const float* __restrict__ bias, const float* __restrict__ comb, int col,
    float* __restrict__ out, int N, int K) {
    __shared__ __bf16 As[2][128 * 32];
    __shared__ __bf16 Bs[2][128 * 32];

    int tid = threadIdx.x;
    int n0 = blockIdx.x * 128, m0 = blockIdx.y * 128;
    int w = tid >> 5, lane = tid & 31;
    int wm = w >> 1, wn = w & 1;
    int lidx = lane & 15, lhalf = lane >> 4;

    v8f acc[2][4];
    #pragma unroll
    for (int i = 0; i < 2; ++i)
        #pragma unroll
        for (int j = 0; j < 4; ++j)
            acc[i][j] = (v8f){0.f,0.f,0.f,0.f,0.f,0.f,0.f,0.f};

    load_a_tile(As[0], A, m0, 0, K, tid);
    load_b_tile(Bs[0], W, n0, 0, N, tid);
    prefetch_tiles(A, W, m0, n0, 32, N, K, tid);
    __syncthreads();

    int KT = K >> 5;
    for (int kt = 0; kt < KT; ++kt) {
        int cur = kt & 1;
        if (kt + 1 < KT) {
            load_a_tile(As[cur ^ 1], A, m0, (kt + 1) << 5, K, tid);
            load_b_tile(Bs[cur ^ 1], W, n0, (kt + 1) << 5, N, tid);
        }
        if (kt + 2 < KT)
            prefetch_tiles(A, W, m0, n0, (kt + 2) << 5, N, K, tid);
        AFrag a[2], b[4];
        #pragma unroll
        for (int i = 0; i < 2; ++i)
            make_frag_a(a[i], As[cur], wm * 32 + i * 16, lidx, lhalf);
        #pragma unroll
        for (int j = 0; j < 4; ++j)
            make_frag_b(b[j], Bs[cur], wn * 64 + j * 16, lidx, lhalf);
        #pragma unroll
        for (int i = 0; i < 2; ++i)
            #pragma unroll
            for (int j = 0; j < 4; ++j)
                acc[i][j] = __builtin_amdgcn_wmma_f32_16x16x32_bf16(
                    false, a[i].v, false, b[j].v, (short)0, acc[i][j],
                    false, false);
        __syncthreads();
    }

    #pragma unroll
    for (int i = 0; i < 2; ++i) {
        int rm = m0 + wm * 32 + i * 16 + lhalf * 8;
        int cn[4]; float bv[4];
        #pragma unroll
        for (int j = 0; j < 4; ++j) {
            cn[j] = n0 + wn * 64 + j * 16 + lidx;
            bv[j] = bias[cn[j]];
        }
        #pragma unroll
        for (int r = 0; r < 8; ++r) {
            int gm = rm + r;
            float s = comb[(size_t)gm * COMB_STRIDE + col];
            #pragma unroll
            for (int j = 0; j < 4; ++j) {
                size_t off = (size_t)gm * N + cn[j];
                out[off] += s * (acc[i][j][r] + bv[j]);
            }
        }
    }
}

// ---------------------------------------------------------------------------
extern "C" void kernel_launch(void* const* d_in, const int* in_sizes, int n_in,
                              void* d_out, int out_size, void* d_ws, size_t ws_size,
                              hipStream_t stream) {
    (void)in_sizes; (void)n_in; (void)out_size; (void)ws_size;
    const float* x      = (const float*)d_in[0];
    const float* gate_w = (const float*)d_in[1];
    const float* w1     = (const float*)d_in[2];
    const float* b1     = (const float*)d_in[3];
    const float* w2     = (const float*)d_in[4];
    const float* b2     = (const float*)d_in[5];
    const float* sw1    = (const float*)d_in[6];
    const float* sb1    = (const float*)d_in[7];
    const float* sw2    = (const float*)d_in[8];
    const float* sb2    = (const float*)d_in[9];
    float* out = (float*)d_out;

    char* ws = (char*)d_ws;
    __bf16* xb   = (__bf16*)ws;                                       // 8 MB
    __bf16* hb   = (__bf16*)(ws + (size_t)T_TOK * HDIM * 2);          // 32 MB
    float*  comb = (float*)(ws + (size_t)T_TOK * HDIM * 2
                               + (size_t)T_TOK * FDIM * 2);           // 144 KB

    hipMemsetAsync(d_out, 0, (size_t)T_TOK * HDIM * sizeof(float), stream);

    moe_convert_bf16<<<(T_TOK * HDIM) / 2048, 256, 0, stream>>>(x, xb, T_TOK * HDIM);
    moe_gate<<<T_TOK / 256, 256, 0, stream>>>(x, gate_w, comb);

    dim3 g1(FDIM / 128, T_TOK / 128);   // (32, 32)
    dim3 g2(HDIM / 128, T_TOK / 128);   // (8, 32)

    for (int e = 0; e < NEXP; ++e) {
        moe_gemm1<<<g1, 256, 0, stream>>>(
            xb, w1 + (size_t)e * HDIM * FDIM, b1 + (size_t)e * FDIM, hb,
            FDIM, HDIM);
        moe_gemm2<<<g2, 256, 0, stream>>>(
            hb, w2 + (size_t)e * FDIM * HDIM, b2 + (size_t)e * HDIM, comb, e,
            out, HDIM, FDIM);
    }
    // shared expert (combine weight column 8 == 1.0)
    moe_gemm1<<<g1, 256, 0, stream>>>(xb, sw1, sb1, hb, FDIM, HDIM);
    moe_gemm2<<<g2, 256, 0, stream>>>(hb, sw2, sb2, comb, NEXP, out, HDIM, FDIM);
}
